// Top2MoE_32530082300118
// MI455X (gfx1250) — compile-verified, matrix operand order checked by reading, not run on hardware
//
#include <hip/hip_runtime.h>
#include <hip/hip_bf16.h>

typedef __attribute__((ext_vector_type(16))) __bf16 v16bf;
typedef __attribute__((ext_vector_type(8)))  __bf16 v8bf;
typedef __attribute__((ext_vector_type(4)))  __bf16 v4bf;
typedef __attribute__((ext_vector_type(8)))  float  v8f;
typedef __attribute__((ext_vector_type(4)))  float  v4f;
typedef int i32x4 __attribute__((vector_size(16)));  // gcc-style vector: matches builtin proto

#define NTOK 16384
#define DDIM 1024
#define NEXP 8

#define TM 128
#define TN 64
#define TK 64
#define LDSP 72          // padded LDS row stride in bf16 (144 B: conflict-free b128 frags)
#define NSTG (DDIM / TK) // 16 K-stages

// ---------- CDNA5 async global->LDS (ASYNCcnt) with guarded fallback ----------
#if defined(__has_builtin)
#if __has_builtin(__builtin_amdgcn_global_load_async_to_lds_b128)
#define HAS_ASYNC_LDS 1
#endif
#endif
#ifndef HAS_ASYNC_LDS
#define HAS_ASYNC_LDS 0
#endif

__device__ __forceinline__ void async_cp16(const __bf16* g, __bf16* l) {
#if HAS_ASYNC_LDS
  typedef __attribute__((address_space(1))) i32x4 gi32x4;  // global AS1 pointee
  typedef __attribute__((address_space(3))) i32x4 li32x4;  // LDS AS3 pointee
  // flat-LDS addr[31:0] == AS3 offset (ISA 10.2); global flat addr is identity.
  __builtin_amdgcn_global_load_async_to_lds_b128(
      (gi32x4*)(uintptr_t)g,
      (li32x4*)(unsigned)(uintptr_t)l,
      0, 0);
#else
  *(v8bf*)l = *(const v8bf*)g;
#endif
}

__device__ __forceinline__ void async_wait0() {
#if HAS_ASYNC_LDS
#if __has_builtin(__builtin_amdgcn_s_wait_asynccnt)
  __builtin_amdgcn_s_wait_asynccnt(0);
#else
  asm volatile("s_wait_asynccnt 0x0" ::: "memory");
#endif
#endif
}

// ---------------- Gate: softmax + top-2 -> per-token (scale0, scale1) ----------------
__global__ __launch_bounds__(256) void gate_kernel(
    const float* __restrict__ tokens, const float* __restrict__ gate_w,
    float2* __restrict__ scales)
{
  const int wave = threadIdx.x >> 5;
  const int lane = threadIdx.x & 31;
  const int tok  = blockIdx.x * 8 + wave;

  const float* x = tokens + (size_t)tok * DDIM;
  float acc[NEXP];
#pragma unroll
  for (int e = 0; e < NEXP; ++e) acc[e] = 0.f;

  for (int d = lane; d < DDIM; d += 32) {
    float xv = x[d];
    const v4f* g = (const v4f*)(gate_w + (size_t)d * NEXP);
    v4f g0 = g[0], g1 = g[1];
    acc[0] += xv * g0[0]; acc[1] += xv * g0[1];
    acc[2] += xv * g0[2]; acc[3] += xv * g0[3];
    acc[4] += xv * g1[0]; acc[5] += xv * g1[1];
    acc[6] += xv * g1[2]; acc[7] += xv * g1[3];
  }
#pragma unroll
  for (int off = 16; off > 0; off >>= 1) {
#pragma unroll
    for (int e = 0; e < NEXP; ++e) acc[e] += __shfl_down(acc[e], off, 32);
  }
  if (lane == 0) {
    float m = acc[0];
#pragma unroll
    for (int e = 1; e < NEXP; ++e) m = fmaxf(m, acc[e]);
    float w[NEXP], s = 0.f;
#pragma unroll
    for (int e = 0; e < NEXP; ++e) { w[e] = __expf(acc[e] - m); s += w[e]; }
    float inv = 1.f / s;
    int i0 = 0; float b0v = acc[0];
#pragma unroll
    for (int e = 1; e < NEXP; ++e) if (acc[e] > b0v) { b0v = acc[e]; i0 = e; }
    int i1 = -1; float b1v = -3.4e38f;
#pragma unroll
    for (int e = 0; e < NEXP; ++e) if (e != i0 && acc[e] > b1v) { b1v = acc[e]; i1 = e; }
    float2 sc;
    sc.x = (i0 == 0) ? w[0] * inv : 0.f;
    sc.y = (i1 == 1) ? w[1] * inv : 0.f;
    scales[tok] = sc;
  }
}

// ---------------- f32 -> bf16 bulk conversion (pre-pass into workspace) ----------------
__global__ __launch_bounds__(256) void cvt_kernel(
    const float* __restrict__ src, __bf16* __restrict__ dst)
{
  size_t i = ((size_t)blockIdx.x * 256 + threadIdx.x) * 8;
  v4f a = *(const v4f*)(src + i);
  v4f b = *(const v4f*)(src + i + 4);
  v8bf o;
#pragma unroll
  for (int j = 0; j < 4; ++j) { o[j] = (__bf16)a[j]; o[j + 4] = (__bf16)b[j]; }
  *(v8bf*)(dst + i) = o;
}

// ---------------- Fused dual-expert GEMM + bias + SiLU + gating ----------------
// XPRE: tokens pre-converted to bf16 in ws (async-stage them)
// WPRE: weights pre-converted to bf16 in ws (async-stage them)
template <bool XPRE, bool WPRE>
__global__ __launch_bounds__(256) void moe_kernel(
    const float* __restrict__ Xf,  const __bf16* __restrict__ Xb,
    const float* __restrict__ W0f, const float* __restrict__ W1f,
    const __bf16* __restrict__ W0b, const __bf16* __restrict__ W1b,
    const float* __restrict__ B0,  const float* __restrict__ B1,
    const float2* __restrict__ scales, float* __restrict__ out)
{
  __shared__ __align__(16) __bf16 Xs[2][TM][LDSP];      // 36 KB
  __shared__ __align__(16) __bf16 Ws[2][2][TN][LDSP];   // 36 KB ([buf][expert])
  __shared__ float s0s[TM], s1s[TM];

  const int tid    = threadIdx.x;
  const int lane   = tid & 31;
  const int waveId = tid >> 5;
  const int wm     = waveId & 3;   // 4 wave-rows of 32
  const int wn     = waveId >> 2;  // 2 wave-cols of 32
  const int half   = lane >> 4;
  const int ln     = lane & 15;

  const int rBase = blockIdx.y * TM;
  const int cBase = blockIdx.x * TN;

  if (tid < TM) {
    float2 sc = scales[rBase + tid];
    s0s[tid] = sc.x;
    s1s[tid] = sc.y;
  }

  v8f acc[2][2][2];  // [expert][mi][ni]
#pragma unroll
  for (int e = 0; e < 2; ++e)
#pragma unroll
    for (int mi = 0; mi < 2; ++mi)
#pragma unroll
      for (int ni = 0; ni < 2; ++ni)
#pragma unroll
        for (int j = 0; j < 8; ++j) acc[e][mi][ni][j] = 0.f;

  const int bRow  = tid >> 3;        // 0..31  (16B-granule staging)
  const int bCol  = (tid & 7) * 8;   // 0..56  (bf16 elements, 16B chunks)
  const int ldRow = tid >> 4;        // 0..15  (f32 fallback staging)
  const int ldCol = (tid & 15) * 4;

  v4f xreg[8];  // live only in !XPRE mode

  auto loadXregs = [&](int kb) {
    if (!XPRE) {
#pragma unroll
      for (int it = 0; it < 8; ++it)
        xreg[it] = *(const v4f*)(Xf + (size_t)(rBase + ldRow + it * 16) * DDIM + kb + ldCol);
    }
  };
  auto storeXregs = [&](int buf) {
    if (!XPRE) {
#pragma unroll
      for (int it = 0; it < 8; ++it) {
        v4bf b;
#pragma unroll
        for (int j = 0; j < 4; ++j) b[j] = (__bf16)xreg[it][j];
        *(v4bf*)&Xs[buf][ldRow + it * 16][ldCol] = b;
      }
    }
  };
  auto issueStage = [&](int kb, int buf) {
    if (WPRE) {
#pragma unroll
      for (int it = 0; it < 2; ++it) {
        int r = bRow + it * 32;  // 0..63
        async_cp16(W0b + (size_t)(cBase + r) * DDIM + kb + bCol, &Ws[buf][0][r][bCol]);
        async_cp16(W1b + (size_t)(cBase + r) * DDIM + kb + bCol, &Ws[buf][1][r][bCol]);
      }
    } else {
#pragma unroll
      for (int it = 0; it < 4; ++it) {
        int r = ldRow + it * 16;
        v4f v0 = *(const v4f*)(W0f + (size_t)(cBase + r) * DDIM + kb + ldCol);
        v4f v1 = *(const v4f*)(W1f + (size_t)(cBase + r) * DDIM + kb + ldCol);
        v4bf c0, c1;
#pragma unroll
        for (int j = 0; j < 4; ++j) { c0[j] = (__bf16)v0[j]; c1[j] = (__bf16)v1[j]; }
        *(v4bf*)&Ws[buf][0][r][ldCol] = c0;
        *(v4bf*)&Ws[buf][1][r][ldCol] = c1;
      }
    }
    if (XPRE) {
#pragma unroll
      for (int it = 0; it < 4; ++it) {
        int r = bRow + it * 32;  // 0..127
        async_cp16(Xb + (size_t)(rBase + r) * DDIM + kb + bCol, &Xs[buf][r][bCol]);
      }
    }
  };
  auto computeStage = [&](int buf) {
#pragma unroll
    for (int kk = 0; kk < TK; kk += 32) {
      const int klo = kk + half * 8;
      const int khi = kk + 16 + half * 8;
      v16bf afr[2];
#pragma unroll
      for (int mi = 0; mi < 2; ++mi) {
        int r = wm * 32 + mi * 16 + ln;
        v8bf lo = *(const v8bf*)&Xs[buf][r][klo];
        v8bf hi = *(const v8bf*)&Xs[buf][r][khi];
        afr[mi] = __builtin_shufflevector(lo, hi, 0,1,2,3,4,5,6,7,8,9,10,11,12,13,14,15);
      }
#pragma unroll
      for (int ni = 0; ni < 2; ++ni) {
        int c = wn * 32 + ni * 16 + ln;
        v8bf lo0 = *(const v8bf*)&Ws[buf][0][c][klo];
        v8bf hi0 = *(const v8bf*)&Ws[buf][0][c][khi];
        v16bf bf0 = __builtin_shufflevector(lo0, hi0, 0,1,2,3,4,5,6,7,8,9,10,11,12,13,14,15);
        v8bf lo1 = *(const v8bf*)&Ws[buf][1][c][klo];
        v8bf hi1 = *(const v8bf*)&Ws[buf][1][c][khi];
        v16bf bf1 = __builtin_shufflevector(lo1, hi1, 0,1,2,3,4,5,6,7,8,9,10,11,12,13,14,15);
#pragma unroll
        for (int mi = 0; mi < 2; ++mi) {
          acc[0][mi][ni] = __builtin_amdgcn_wmma_f32_16x16x32_bf16(
              false, afr[mi], false, bf0, (short)0, acc[0][mi][ni], false, false);
          acc[1][mi][ni] = __builtin_amdgcn_wmma_f32_16x16x32_bf16(
              false, afr[mi], false, bf1, (short)0, acc[1][mi][ni], false, false);
        }
      }
    }
  };

  // -------- double-buffered pipeline --------
  loadXregs(0);
  issueStage(0, 0);
  storeXregs(0);
  async_wait0();
  __syncthreads();

  for (int s = 0; s < NSTG; ++s) {
    const int cur = s & 1, nxt = cur ^ 1;
    if (s + 1 < NSTG) {
      loadXregs((s + 1) * TK);        // global loads in flight during WMMA
      issueStage((s + 1) * TK, nxt);  // async fills in flight during WMMA
    }
    computeStage(cur);
    if (s + 1 < NSTG) storeXregs(nxt);
    async_wait0();
    __syncthreads();
  }

  // -------- epilogue: bias + SiLU + gate scaling --------
#pragma unroll
  for (int ni = 0; ni < 2; ++ni) {
    int c = cBase + wn * 32 + ni * 16 + ln;
    float bb0 = B0[c], bb1 = B1[c];
#pragma unroll
    for (int mi = 0; mi < 2; ++mi) {
#pragma unroll
      for (int v = 0; v < 8; ++v) {
        int rl = wm * 32 + mi * 16 + half * 8 + v;
        float a0 = acc[0][mi][ni][v] + bb0;
        float a1 = acc[1][mi][ni][v] + bb1;
        float e0 = a0 / (1.f + __expf(-a0));
        float e1 = a1 / (1.f + __expf(-a1));
        out[(size_t)(rBase + rl) * DDIM + c] = e0 * s0s[rl] + e1 * s1s[rl];
      }
    }
  }
}

extern "C" void kernel_launch(void* const* d_in, const int* in_sizes, int n_in,
                              void* d_out, int out_size, void* d_ws, size_t ws_size,
                              hipStream_t stream) {
  (void)in_sizes; (void)n_in; (void)out_size;
  const float* tokens = (const float*)d_in[0];
  const float* gate_w = (const float*)d_in[1];
  const float* w0     = (const float*)d_in[2];
  const float* b0     = (const float*)d_in[3];
  const float* w1     = (const float*)d_in[4];
  const float* b1     = (const float*)d_in[5];
  float* out = (float*)d_out;

  // workspace layout: [scales 128KB][w0 bf16 2MB][w1 bf16 2MB][tokens bf16 32MB]
  const size_t offW  = 128 * 1024;
  const size_t offX  = offW + 2 * (size_t)DDIM * DDIM * sizeof(__bf16);
  const size_t endX  = offX + (size_t)NTOK * DDIM * sizeof(__bf16);
  const bool wpre = ws_size >= offX;
  const bool xpre = ws_size >= endX;

  float2* scales = (float2*)d_ws;
  __bf16* w0b = (__bf16*)((char*)d_ws + offW);
  __bf16* w1b = w0b + (size_t)DDIM * DDIM;
  __bf16* xb  = (__bf16*)((char*)d_ws + offX);

  gate_kernel<<<NTOK / 8, 256, 0, stream>>>(tokens, gate_w, scales);
  if (wpre) {
    cvt_kernel<<<(DDIM * DDIM) / 2048, 256, 0, stream>>>(w0, w0b);
    cvt_kernel<<<(DDIM * DDIM) / 2048, 256, 0, stream>>>(w1, w1b);
  }
  if (xpre) {
    cvt_kernel<<<(NTOK * DDIM) / 2048, 256, 0, stream>>>(tokens, xb);
  }

  dim3 grid(DDIM / TN, NTOK / TM);  // (16, 128)
  if (xpre && wpre) {
    moe_kernel<true, true><<<grid, 256, 0, stream>>>(
        tokens, xb, w0, w1, w0b, w1b, b0, b1, scales, out);
  } else if (wpre) {
    moe_kernel<false, true><<<grid, 256, 0, stream>>>(
        tokens, xb, w0, w1, w0b, w1b, b0, b1, scales, out);
  } else {
    moe_kernel<false, false><<<grid, 256, 0, stream>>>(
        tokens, xb, w0, w1, w0b, w1b, b0, b1, scales, out);
  }
}
